// pc_encoder_74818330296426
// MI455X (gfx1250) — matrix operand order repaired and checked by reading, not compiled, not run on hardware
//
#include <hip/hip_runtime.h>
#include <hip/hip_bf16.h>

typedef __attribute__((ext_vector_type(16))) _Float16 v16h;
typedef __attribute__((ext_vector_type(8)))  float    v8f;
typedef unsigned int u32x4 __attribute__((ext_vector_type(4)));
typedef int          i32x4 __attribute__((ext_vector_type(4)));
typedef int          i32x8 __attribute__((ext_vector_type(8)));

#define BP 8
#define NPTS 8192
#define NS 1024
#define CIN 6
#define KPAD 200   // LDS row stride (halves) for W panel (max IC = 192)
#define XPAD 34    // LDS row stride (halves) for transposed X tile
#define STATN 128  // max conv channels

#if defined(__has_builtin)
# if __has_builtin(__builtin_amdgcn_tensor_load_to_lds)
#  define HAVE_TDM 1
# endif
#endif
#ifndef HAVE_TDM
# define HAVE_TDM 0
#endif
#if __has_include(<hip/amd_detail/amd_gfx1250_TDM.h>)
# define TDM_6ARG 1
#else
# define TDM_6ARG 0
#endif

// ---------------- reductions ----------------
__device__ inline float wave_sum_f(float v) {
  for (int o = 16; o > 0; o >>= 1) v += __shfl_down(v, o);
  return v;
}
__device__ inline int wave_sum_i(int v) {
  for (int o = 16; o > 0; o >>= 1) v += __shfl_down(v, o);
  return v;
}

#if HAVE_TDM
// 2D f32 tile load via Tensor Data Mover (D# per CDNA5 ISA)
__device__ __forceinline__ void tdm_load_2d_f32(const float* gptr, unsigned lds_off,
                                                unsigned tile0, unsigned tile1,
                                                unsigned tdim0, unsigned tdim1,
                                                unsigned stride0) {
  unsigned long long ga = (unsigned long long)(uintptr_t)gptr;
  u32x4 g0;
  g0[0] = 1u;                                        // count=1, user mode
  g0[1] = lds_off;                                   // lds_addr
  g0[2] = (unsigned)(ga & 0xffffffffu);              // global_addr[31:0]
  g0[3] = (unsigned)((ga >> 32) & 0x01ffffffu) | (2u << 30);  // addr[56:32] | type=2
  i32x8 g1;
  g1[0] = (int)(2u << 16);                           // data_size = 4B
  g1[1] = (int)((tdim0 & 0xffffu) << 16);            // tensor_dim0[15:0]
  g1[2] = (int)(((tdim0 >> 16) & 0xffffu) | ((tdim1 & 0xffffu) << 16));
  g1[3] = (int)(((tdim1 >> 16) & 0xffffu) | ((tile0 & 0xffffu) << 16)); // tile_dim0
  g1[4] = (int)(tile1 & 0xffffu);                    // tile_dim1 (tile_dim2 = 0)
  g1[5] = (int)stride0;                              // tensor_dim0_stride[31:0]
  g1[6] = 0;
  g1[7] = 0;
  i32x4 gz = {0, 0, 0, 0};
#if TDM_6ARG
  i32x8 gz8 = {0, 0, 0, 0, 0, 0, 0, 0};
  __builtin_amdgcn_tensor_load_to_lds(g0, g1, gz, gz, gz8, 0);
#else
  __builtin_amdgcn_tensor_load_to_lds(g0, g1, gz, gz, 0);
#endif
}
#endif

// ---------------- binary mask (forward value == hard mask) ----------------
__global__ void mask_kernel(const float* __restrict__ aff, float* __restrict__ bm,
                            int n, int kmin) {
  int b = blockIdx.x;
  const float* a = aff + (long)b * n;
  float* m = bm + (long)b * n;
  __shared__ int sC[32];
  __shared__ int sTot;
  int t = threadIdx.x, nw = blockDim.x >> 5;

  int cnt = 0;
  for (int i = t; i < n; i += blockDim.x) cnt += (a[i] > 0.5f) ? 1 : 0;
  cnt = wave_sum_i(cnt);
  if ((t & 31) == 0) sC[t >> 5] = cnt;
  __syncthreads();
  if (t == 0) { int tot = 0; for (int w = 0; w < nw; ++w) tot += sC[w]; sTot = tot; }
  __syncthreads();
  float thr = 0.5f;
  if (sTot == 0) {
    float lo = 0.f, hi = 1.f;
    for (int it = 0; it < 30; ++it) {
      float mid = 0.5f * (lo + hi);
      int c = 0;
      for (int i = t; i < n; i += blockDim.x) c += (a[i] >= mid) ? 1 : 0;
      c = wave_sum_i(c);
      __syncthreads();
      if ((t & 31) == 0) sC[t >> 5] = c;
      __syncthreads();
      if (t == 0) { int tot = 0; for (int w = 0; w < nw; ++w) tot += sC[w]; sTot = tot; }
      __syncthreads();
      if (sTot >= kmin) lo = mid; else hi = mid;
    }
    thr = lo;
  }
  for (int i = t; i < n; i += blockDim.x) m[i] = (a[i] >= thr) ? 1.f : 0.f;
}

// ---------------- farthest point sampling ----------------
__global__ void fps_kernel(const float* __restrict__ pts, int* __restrict__ idxout) {
  int b = blockIdx.x;
  const float* P = pts + (long)b * NPTS * CIN;
  int t = threadIdx.x;              // 1024 threads
  const int PPT = NPTS / 1024;      // 8 points/thread
  float px[8], py[8], pz[8], dd[8];
  for (int j = 0; j < PPT; ++j) {
    int p = t + 1024 * j;
    px[j] = P[p * CIN + 0]; py[j] = P[p * CIN + 1]; pz[j] = P[p * CIN + 2];
    dd[j] = 1e10f;
  }
  __shared__ float sFx, sFy, sFz;
  __shared__ int sFar;
  __shared__ float sV[32];
  __shared__ int sI[32];
  int far = 0;
  for (int step = 0; step < NS; ++step) {
    if (t == 0) {
      idxout[b * NS + step] = far;
      sFx = P[far * CIN + 0]; sFy = P[far * CIN + 1]; sFz = P[far * CIN + 2];
    }
    __syncthreads();
    float fx = sFx, fy = sFy, fz = sFz;
    float best = -1.f; int bi = 0x7fffffff;
    for (int j = 0; j < PPT; ++j) {
      float dx = px[j] - fx, dy = py[j] - fy, dz = pz[j] - fz;
      float d = dx * dx + dy * dy + dz * dz;
      dd[j] = fminf(dd[j], d);
      int p = t + 1024 * j;
      if (dd[j] > best || (dd[j] == best && p < bi)) { best = dd[j]; bi = p; }
    }
    for (int o = 16; o > 0; o >>= 1) {
      float ov = __shfl_down(best, o);
      int   oi = __shfl_down(bi, o);
      if (ov > best || (ov == best && oi < bi)) { best = ov; bi = oi; }
    }
    if ((t & 31) == 0) { sV[t >> 5] = best; sI[t >> 5] = bi; }
    __syncthreads();
    if (t == 0) {
      float bv = sV[0]; int bb = sI[0];
      for (int w = 1; w < 32; ++w)
        if (sV[w] > bv || (sV[w] == bv && sI[w] < bb)) { bv = sV[w]; bb = sI[w]; }
      sFar = bb;
    }
    __syncthreads();
    far = sFar;
  }
}

// ---------------- gather sampled points + aff ----------------
__global__ void gather_sampled(const float* __restrict__ pts, const float* __restrict__ aff,
                               const int* __restrict__ fidx, float* __restrict__ sampled,
                               float* __restrict__ saff, float* __restrict__ out_saff) {
  int b = blockIdx.y;
  int s = blockIdx.x * blockDim.x + threadIdx.x;
  if (s >= NS) return;
  int p = fidx[b * NS + s];
  for (int c = 0; c < CIN; ++c)
    sampled[((long)b * NS + s) * CIN + c] = pts[((long)b * NPTS + p) * CIN + c];
  float a = aff[(long)b * NPTS + p];
  saff[b * NS + s] = a;
  out_saff[b * NS + s] = a;
}

// ---------------- sum of bmask_s ----------------
__global__ void msum_kernel(const float* __restrict__ bm, float* __restrict__ out) {
  __shared__ float sv[32];
  int t = threadIdx.x;
  float s = 0.f;
  for (int i = t; i < BP * NS; i += 1024) s += bm[i];
  s = wave_sum_f(s);
  if ((t & 31) == 0) sv[t >> 5] = s;
  __syncthreads();
  if (t == 0) { float tot = 0.f; for (int w = 0; w < 32; ++w) tot += sv[w]; *out = tot; }
}

// ---------------- KNN (k = 32; LDS-resident distance row) ----------------
__global__ void knn_kernel(const float* __restrict__ pts, const float* __restrict__ sampled,
                           const float* __restrict__ bmask, const float* __restrict__ bmask_s,
                           int* __restrict__ idxout, int masked) {
  int b = blockIdx.y, s = blockIdx.x;
  __shared__ float d[NPTS];
  __shared__ float sv[8];
  __shared__ int si[8];
  int t = threadIdx.x; // 256
  float qx = sampled[((long)b * NS + s) * CIN + 0];
  float qy = sampled[((long)b * NS + s) * CIN + 1];
  float qz = sampled[((long)b * NS + s) * CIN + 2];
  if (masked) { float m = bmask_s[b * NS + s]; qx *= m; qy *= m; qz *= m; }
  for (int i = t; i < NPTS; i += 256) {
    float x = pts[((long)b * NPTS + i) * CIN + 0];
    float y = pts[((long)b * NPTS + i) * CIN + 1];
    float z = pts[((long)b * NPTS + i) * CIN + 2];
    if (masked && bmask[(long)b * NPTS + i] == 0.f) { x = 1e9f; y = 1e9f; z = 1e9f; }
    float dx = qx - x, dy = qy - y, dz = qz - z;
    d[i] = dx * dx + dy * dy + dz * dz;
  }
  __syncthreads();
  for (int it = 0; it < 32; ++it) {
    float best = 3.3e38f; int bi = 0x7fffffff;
    for (int i = t; i < NPTS; i += 256) {
      float v = d[i];
      if (v < best || (v == best && i < bi)) { best = v; bi = i; }
    }
    for (int o = 16; o > 0; o >>= 1) {
      float ov = __shfl_down(best, o);
      int   oi = __shfl_down(bi, o);
      if (ov < best || (ov == best && oi < bi)) { best = ov; bi = oi; }
    }
    if ((t & 31) == 0) { sv[t >> 5] = best; si[t >> 5] = bi; }
    __syncthreads();
    if (t == 0) {
      float bv = sv[0]; int bb = si[0];
      for (int w = 1; w < 8; ++w)
        if (sv[w] < bv || (sv[w] == bv && si[w] < bb)) { bv = sv[w]; bb = si[w]; }
      idxout[((long)b * NS + s) * 32 + it] = bb;
      d[bb] = 3.4e38f;
    }
    __syncthreads();
  }
}

// ---------------- grouped gather -> [b][c][s*K+j] ----------------
__global__ void group_gather(const float* __restrict__ pts, const float* __restrict__ bmask,
                             const int* __restrict__ idx, float* __restrict__ g,
                             int K, int masked) {
  int b = blockIdx.y;
  int col = blockIdx.x * blockDim.x + threadIdx.x;
  int NCOL = NS * K;
  if (col >= NCOL) return;
  int s = col / K, j = col % K;
  int p = idx[((long)b * NS + s) * 32 + j];
  float m = masked ? bmask[(long)b * NPTS + p] : 1.f;
  for (int c = 0; c < CIN; ++c)
    g[((long)b * CIN + c) * NCOL + col] = pts[((long)b * NPTS + p) * CIN + c] * m;
}

// ---------------- WMMA GEMM: Y = W(OCxIC) * X(ICxNCOL) + bias ----------------
// 128 threads = 4 waves. WG tile: M=16, N=128; each wave computes two 16x16
// f32 tiles via v_wmma_f32_16x16x32_f16. W panel DMA'd to LDS by the TDM.
__global__ void wmma_gemm(const float* __restrict__ X, const float* __restrict__ W,
                          const float* __restrict__ bias, float* __restrict__ Y,
                          int IC, int OC, int NCOL,
                          long xStrideB, long yStrideB, long yStrideO, long yStrideC,
                          int wColMajor,
                          const float* __restrict__ colmask, int Kgrp,
                          const float* __restrict__ outmask, int SM) {
  __shared__ _Float16 sW[16 * KPAD];
  __shared__ _Float16 sXT[128][XPAD];   // transposed X tile: [col][k]
  __shared__ float sWstage[16 * 192];   // TDM staging (f32)
  int b = blockIdx.z;
  int obase = blockIdx.y * 16;
  int colbase = blockIdx.x * 128;
  int t = threadIdx.x;
  int wave = t >> 5, lane = t & 31;

  for (int i = t; i < 16 * KPAD; i += 128) sW[i] = (_Float16)0.f;
  __syncthreads();

#if HAVE_TDM
  if (t < 32) {
    unsigned lds_off = (unsigned)(uintptr_t)&sWstage[0];
    if (wColMajor)   // W[(c,o)] row-major: tile = 16 contiguous o's x IC rows
      tdm_load_2d_f32(W + obase, lds_off, 16u, (unsigned)IC,
                      (unsigned)(OC - obase), (unsigned)IC, (unsigned)OC);
    else             // W[(o,c)] row-major: tile = IC contiguous c's x 16 rows
      tdm_load_2d_f32(W + (long)obase * IC, lds_off, (unsigned)IC, 16u,
                      (unsigned)IC, (unsigned)(OC - obase), (unsigned)IC);
# if __has_builtin(__builtin_amdgcn_s_wait_tensorcnt)
    __builtin_amdgcn_s_wait_tensorcnt(0);
# endif
  }
  __syncthreads();
  for (int i = t; i < 16 * IC; i += 128) {
    int m = i / IC, c = i - m * IC;
    float wv = wColMajor ? sWstage[c * 16 + m] : sWstage[m * IC + c];
    sW[m * KPAD + c] = (_Float16)wv;
  }
#else
  for (int i = t; i < 16 * IC; i += 128) {
    int m = i / IC, c = i - m * IC;
    int oo = obase + m;
    float wv = 0.f;
    if (oo < OC) wv = wColMajor ? W[(long)c * OC + oo] : W[(long)oo * IC + c];
    sW[m * KPAD + c] = (_Float16)wv;
  }
#endif
  __syncthreads();

  v8f acc0 = {0.f, 0.f, 0.f, 0.f, 0.f, 0.f, 0.f, 0.f};
  v8f acc1 = {0.f, 0.f, 0.f, 0.f, 0.f, 0.f, 0.f, 0.f};
  int hi = lane >> 4;
  int m = lane & 15;

  for (int kb = 0; kb < IC; kb += 32) {
    __syncthreads();
    // load 32 x 128 X tile, transposed into LDS (packed pairs along K)
    for (int i = t; i < 16 * 128; i += 128) {
      int col = i & 127;
      int kp = i >> 7;                 // 0..15 -> k = 2*kp, 2*kp+1
      int gcol = colbase + col;
      float v0 = 0.f, v1 = 0.f;
      if (gcol < NCOL) {
        int c0 = kb + 2 * kp;
        float cm = colmask ? colmask[b * SM + gcol / Kgrp] : 1.f;
        if (c0 < IC)     v0 = X[(long)b * xStrideB + (long)c0 * NCOL + gcol] * cm;
        if (c0 + 1 < IC) v1 = X[(long)b * xStrideB + (long)(c0 + 1) * NCOL + gcol] * cm;
      }
      sXT[col][2 * kp]     = (_Float16)v0;
      sXT[col][2 * kp + 1] = (_Float16)v1;
    }
    if (kb + 32 < IC)
      __builtin_prefetch(&X[(long)b * xStrideB + (long)(kb + 32) * NCOL + colbase + t], 0, 0);
    __syncthreads();

    union { v16h v; _Float16 h[16]; } A, B0, B1;
#pragma unroll
    for (int i = 0; i < 8; ++i) {
      A.h[i]     = sW[m * KPAD + kb + hi * 16 + i];       // K = hi*16 + i
      A.h[8 + i] = sW[m * KPAD + kb + hi * 16 + 8 + i];   // K = hi*16 + 8 + i
    }
    int n0 = wave * 32 + (lane & 15);
#pragma unroll
    for (int i = 0; i < 16; ++i) {
      B0.h[i] = sXT[n0][hi * 16 + i];        // contiguous -> ds_load_b128 x2
      B1.h[i] = sXT[n0 + 16][hi * 16 + i];
    }
    acc0 = __builtin_amdgcn_wmma_f32_16x16x32_f16(false, A.v, false, B0.v,
                                                  (short)0, acc0, false, false);
    acc1 = __builtin_amdgcn_wmma_f32_16x16x32_f16(false, A.v, false, B1.v,
                                                  (short)0, acc1, false, false);
  }

  int n2 = lane & 15;
#pragma unroll
  for (int tt = 0; tt < 2; ++tt) {
    int col = colbase + wave * 32 + tt * 16 + n2;
    if (col >= NCOL) continue;
    float om = 1.f;
    if (outmask) om = outmask[b * SM + col / Kgrp];
    v8f acc = tt ? acc1 : acc0;
#pragma unroll
    for (int v = 0; v < 8; ++v) {
      int m2 = (lane < 16) ? v : v + 8;
      int o = obase + m2;
      if (o < OC) {
        float r = acc[v] + (bias ? bias[o] : 0.f);
        Y[(long)b * yStrideB + (long)o * yStrideO + (long)col * yStrideC] = r * om;
      }
    }
  }
}

// ---------------- BN stats / apply ----------------
__global__ void zero_stats(float* __restrict__ stats) {
  stats[threadIdx.x] = 0.f;  // 256 threads -> [0..2*STATN)
}

__global__ void bn_stats(const float* __restrict__ X, float* __restrict__ stats,
                         int NCOL, long xStrideB,
                         const float* __restrict__ colmask, int Kgrp, int SM) {
  int o = blockIdx.x, b = blockIdx.y, t = threadIdx.x; // 256
  const float* row = X + (long)b * xStrideB + (long)o * NCOL;
  float s = 0.f, s2 = 0.f;
  for (int i = t; i < NCOL; i += 256) {
    float v = row[i];
    float mm = colmask ? colmask[b * SM + i / Kgrp] : 1.f;
    s += v * mm; s2 += v * v * mm;
  }
  s = wave_sum_f(s); s2 = wave_sum_f(s2);
  if ((t & 31) == 0) {
    atomicAdd(&stats[o], s);
    atomicAdd(&stats[STATN + o], s2);
  }
}

__global__ void bn_apply(const float* __restrict__ Xin, float* __restrict__ Xout,
                         const float* __restrict__ g, const float* __restrict__ stats,
                         const float* __restrict__ bnw, const float* __restrict__ bnb,
                         int OC, int NCOL, long strideIn, long strideOut,
                         int copyG, int masked, const float* __restrict__ msumPtr, int Kgrp) {
  int b = blockIdx.z, o = blockIdx.y;
  int col = blockIdx.x * 256 + threadIdx.x;
  if (col >= NCOL) return;
  float mean, var;
  if (!masked) {
    float cnt = (float)BP * (float)NCOL;
    mean = stats[o] / cnt;
    var = stats[STATN + o] / cnt - mean * mean;
  } else {
    float msum = *msumPtr;
    float valid = msum + 1e-5f;
    float Mtot = msum * (float)Kgrp;
    float sm = stats[o], s2 = stats[STATN + o];
    mean = sm / valid;
    var = (s2 - 2.f * mean * sm + mean * mean * Mtot) / valid;
  }
  float v = Xin[(long)b * strideIn + (long)o * NCOL + col];
  float r = (v - mean) * rsqrtf(var + 1e-5f) * bnw[o] + bnb[o];
  r = fmaxf(r, 0.f);
  Xout[(long)b * strideOut + (long)o * NCOL + col] = r;
  if (copyG && o < CIN)
    Xout[(long)b * strideOut + (long)(OC + o) * NCOL + col] =
        g[((long)b * CIN + o) * NCOL + col];
}

// ---------------- max-pool over K neighbors ----------------
__global__ void maxpool(const float* __restrict__ X, float* __restrict__ feats,
                        int OC, int K, long strideB, int chOff) {
  int b = blockIdx.z, s = blockIdx.x;
  int o = blockIdx.y * blockDim.x + threadIdx.x;
  if (o >= OC) return;
  const float* row = X + (long)b * strideB + (long)o * (NS * K) + (long)s * K;
  float m = row[0];
  for (int j = 1; j < K; ++j) m = fmaxf(m, row[j]);
  feats[((long)b * 192 + chOff + o) * NS + s] = m;
}

// ---------------- host driver ----------------
extern "C" void kernel_launch(void* const* d_in, const int* in_sizes, int n_in,
                              void* d_out, int out_size, void* d_ws, size_t ws_size,
                              hipStream_t stream) {
  (void)in_sizes; (void)n_in; (void)out_size; (void)ws_size;
  const float* points = (const float*)d_in[0];
  const float* aff    = (const float*)d_in[1];
  // params pytree (dict keys sorted): bn_b[6], bn_w[6], conv_b[6], conv_w[6], embed_b, embed_w
  const float* bn_b_[6]; const float* bn_w_[6]; const float* conv_b_[6]; const float* conv_w_[6];
  for (int i = 0; i < 6; ++i) {
    bn_b_[i]   = (const float*)d_in[2 + i];
    bn_w_[i]   = (const float*)d_in[8 + i];
    conv_b_[i] = (const float*)d_in[14 + i];
    conv_w_[i] = (const float*)d_in[20 + i];
  }
  const float* embed_b = (const float*)d_in[26];
  const float* embed_w = (const float*)d_in[27];

  float* outf = (float*)d_out;
  float* out_pf   = outf;                       // (4,2,1024,512)
  float* out_af   = outf + 4194304;             // (4,2,1024,512)
  float* out_saff = outf + 8388608;             // (4,2,1024,1)

  // workspace layout (floats)
  float* wsf = (float*)d_ws;
  size_t off = 0;
  float* bmask   = wsf + off; off += (size_t)BP * NPTS;
  int*   fpsidx  = (int*)(wsf + off); off += (size_t)BP * NS;
  float* sampled = wsf + off; off += (size_t)BP * NS * CIN;
  float* saff    = wsf + off; off += (size_t)BP * NS;
  float* bmask_s = wsf + off; off += (size_t)BP * NS;
  int*   idxp    = (int*)(wsf + off); off += (size_t)BP * NS * 32;
  int*   idxm    = (int*)(wsf + off); off += (size_t)BP * NS * 32;
  float* gp      = wsf + off; off += (size_t)BP * CIN * NS * 32;
  float* ga      = wsf + off; off += (size_t)BP * CIN * NS * 32;
  float* actA    = wsf + off; off += (size_t)BP * 72 * NS * 32;
  float* actB    = wsf + off; off += (size_t)BP * 72 * NS * 32;
  float* pfeats  = wsf + off; off += (size_t)BP * 192 * NS;
  float* afeats  = wsf + off; off += (size_t)BP * 192 * NS;
  float* stats   = wsf + off; off += 512;  // [0..128) sum, [128..256) sumsq, [256] msum

  // 1. masks, FPS, sampling
  mask_kernel<<<BP, 1024, 0, stream>>>(aff, bmask, NPTS, 819);
  fps_kernel<<<BP, 1024, 0, stream>>>(points, fpsidx);
  gather_sampled<<<dim3(NS / 256, BP), 256, 0, stream>>>(points, aff, fpsidx,
                                                         sampled, saff, out_saff);
  mask_kernel<<<BP, 1024, 0, stream>>>(saff, bmask_s, NS, 102);
  msum_kernel<<<1, 1024, 0, stream>>>(bmask_s, stats + 256);

  // 2. KNN (k=32; k=16 reuses prefix)
  knn_kernel<<<dim3(NS, BP), 256, 0, stream>>>(points, sampled, bmask, bmask_s, idxp, 0);
  knn_kernel<<<dim3(NS, BP), 256, 0, stream>>>(points, sampled, bmask, bmask_s, idxm, 1);

  // 3. MLP blocks
  const int KNNs[2] = {16, 32};
  const int MLPc[2][3] = {{32, 32, 64}, {64, 64, 128}};
  for (int blk = 0; blk < 2; ++blk) {
    int K = KNNs[blk];
    int NCOL = NS * K;
    dim3 ggrid((NCOL + 255) / 256, BP);
    group_gather<<<ggrid, 256, 0, stream>>>(points, bmask, idxp, gp, K, 0);
    group_gather<<<ggrid, 256, 0, stream>>>(points, bmask, idxm, ga, K, 1);

    for (int path = 0; path < 2; ++path) {
      const float* g = path ? ga : gp;
      const float* cmask = path ? bmask_s : nullptr;
      const float* Xin = g;
      long xsB = (long)CIN * NCOL;
      int IC = CIN;
      for (int l = 0; l < 3; ++l) {
        int OC = MLPc[blk][l];
        int pi = blk * 3 + l;
        dim3 gg(NCOL / 128, (OC + 15) / 16, BP);
        wmma_gemm<<<gg, 128, 0, stream>>>(Xin, conv_w_[pi], conv_b_[pi], actA,
                                          IC, OC, NCOL, xsB,
                                          (long)72 * NCOL, (long)NCOL, 1L,
                                          0, cmask, K, nullptr, NS);
        zero_stats<<<1, 256, 0, stream>>>(stats);
        bn_stats<<<dim3(OC, BP), 256, 0, stream>>>(actA, stats, NCOL,
                                                   (long)72 * NCOL, cmask, K, NS);
        bn_apply<<<dim3((NCOL + 255) / 256, OC, BP), 256, 0, stream>>>(
            actA, actB, g, stats, bn_w_[pi], bn_b_[pi], OC, NCOL,
            (long)72 * NCOL, (long)72 * NCOL, (l < 2) ? 1 : 0, path,
            stats + 256, K);
        Xin = actB;
        xsB = (long)72 * NCOL;
        IC = OC + CIN;
      }
      int OCl = MLPc[blk][2];
      maxpool<<<dim3(NS, (OCl + 63) / 64, BP), 64, 0, stream>>>(
          actB, path ? afeats : pfeats, OCl, K, (long)72 * NCOL, blk ? 64 : 0);
    }
  }

  // 4. embedding GEMMs (192 -> 512), output layout (b, s, e)
  dim3 ge(NS / 128, 512 / 16, BP);
  wmma_gemm<<<ge, 128, 0, stream>>>(pfeats, embed_w, embed_b, out_pf,
                                    192, 512, NS, (long)192 * NS,
                                    (long)NS * 512, 1L, 512L,
                                    1, nullptr, 1, nullptr, NS);
  wmma_gemm<<<ge, 128, 0, stream>>>(afeats, embed_w, embed_b, out_af,
                                    192, 512, NS, (long)192 * NS,
                                    (long)NS * 512, 1L, 512L,
                                    1, nullptr, 1, bmask_s, NS);
}
